// PDNL_23888608100550
// MI455X (gfx1250) — compile-verified, matrix operand order checked by reading, not run on hardware
//
#include <hip/hip_runtime.h>
#include <hip/hip_bf16.h>

// MI455X / gfx1250 implementation of PDNL.
// Roofline: epilogue (x + upsample) moves ~540MB => ~23us floor @23.3TB/s.
// Attention GEMMs (2.8 GFLOP) run on v_wmma_f32_16x16x32_f16, fully fused so
// the (N,8192,337) tensors never touch HBM (softmax chain lives in LDS).

typedef __attribute__((ext_vector_type(16))) _Float16 v16h;
typedef __attribute__((ext_vector_type(8)))  float    v8f;

#define NK  337
#define NKP 352
#define NB  8      // batch
#define CH  64
#define CHH 32
#define P4  8192   // 128*64 queries
#define EPSV 1e-6f

__device__ __forceinline__ int imin(int a, int b) { return a < b ? a : b; }

// ---------------- s1[o] = sum_j dd1_w[o,j] ----------------
__global__ __launch_bounds__(256) void k_s1(const float* __restrict__ dd1_w,
                                            float* __restrict__ s1) {
  __shared__ float red[256];
  int o = blockIdx.x, t = threadIdx.x;
  float acc = 0.f;
  for (int j = t; j < 2048; j += 256) acc += dd1_w[(size_t)o * 2048 + j];
  red[t] = acc; __syncthreads();
  for (int st = 128; st > 0; st >>= 1) { if (t < st) red[t] += red[t + st]; __syncthreads(); }
  if (t == 0) s1[o] = red[0];
}

// ---------------- depth2 (bilinear AC to 64x32) -> d2e = depth2 @ dd2_w^T + b ----------------
__global__ __launch_bounds__(256) void k_d2e(const float* __restrict__ dm0,
                                             const float* __restrict__ dd2_w,
                                             const float* __restrict__ dd2_b,
                                             float* __restrict__ d2e) {
  __shared__ float d2[2048];
  int n = blockIdx.x, t = threadIdx.x;
  const float* dm = dm0 + (size_t)n * 512 * 256;
  const float sch = 511.0f / 63.0f, scw = 255.0f / 31.0f;
  for (int idx = t; idx < 2048; idx += 256) {
    int ih = idx >> 5, iw = idx & 31;
    float ph = ih * sch, pw = iw * scw;
    int lh = (int)ph; float fh = ph - lh; lh = imin(lh, 511); int hh2 = imin(lh + 1, 511);
    int lw = (int)pw; float fw = pw - lw; lw = imin(lw, 255); int hw2 = imin(lw + 1, 255);
    d2[idx] = dm[lh * 256 + lw] * (1 - fh) * (1 - fw) + dm[lh * 256 + hw2] * (1 - fh) * fw
            + dm[hh2 * 256 + lw] * fh * (1 - fw)      + dm[hh2 * 256 + hw2] * fh * fw;
  }
  __syncthreads();
  for (int o = t; o < NK; o += 256) {
    float acc = dd2_b[o];
    const float* wr = dd2_w + (size_t)o * 2048;
    for (int j = 0; j < 2048; ++j) acc += d2[j] * wr[j];
    d2e[n * NK + o] = acc;
  }
}

// ---------------- depth1: bilinear AC to 128x64 ----------------
__global__ __launch_bounds__(256) void k_depth1(const float* __restrict__ dm0,
                                                float* __restrict__ depth1) {
  int gid = blockIdx.x * 256 + threadIdx.x;   // N*P4 = 65536
  int p = gid & (P4 - 1), n = gid >> 13;
  int ih = p >> 6, iw = p & 63;
  const float* dm = dm0 + (size_t)n * 512 * 256;
  const float sch = 511.0f / 127.0f, scw = 255.0f / 63.0f;
  float ph = ih * sch, pw = iw * scw;
  int lh = (int)ph; float fh = ph - lh; lh = imin(lh, 511); int hh2 = imin(lh + 1, 511);
  int lw = (int)pw; float fw = pw - lw; lw = imin(lw, 255); int hw2 = imin(lw + 1, 255);
  depth1[gid] = dm[lh * 256 + lw] * (1 - fh) * (1 - fw) + dm[lh * 256 + hw2] * (1 - fh) * fw
              + dm[hh2 * 256 + lw] * fh * (1 - fw)      + dm[hh2 * 256 + hw2] * fh * fw;
}

// ---------------- depthwise 4x4 stride-4 conv ----------------
__global__ __launch_bounds__(256) void k_down(const float* __restrict__ x,
                                              const float* __restrict__ dw,
                                              float* __restrict__ xd) {
  __shared__ float cs[256];
  int b = blockIdx.x;                 // n*64*128 + c*128 + ph
  int ph = b & 127, c = (b >> 7) & 63, n = b >> 13;
  int t = threadIdx.x;                // column 0..255
  const float* src = x + (((size_t)(n * CH + c) * 512) + ph * 4) * 256 + t;
  int s4 = t & 3;
  float acc = 0.f;
#pragma unroll
  for (int r = 0; r < 4; ++r) acc += dw[c * 16 + r * 4 + s4] * src[(size_t)r * 256];
  cs[t] = acc; __syncthreads();
  if (t < 64)
    xd[(((size_t)(n * CH + c) * 128) + ph) * 64 + t] =
        cs[4 * t] + cs[4 * t + 1] + cs[4 * t + 2] + cs[4 * t + 3];
}

// ---------------- fused 1x1 projections: theta(f16), phi_feat, g_feat, atten ----------------
__global__ __launch_bounds__(256) void k_proj(const float* __restrict__ xd,
    const float* __restrict__ tw, const float* __restrict__ tb,
    const float* __restrict__ pw, const float* __restrict__ pb,
    const float* __restrict__ gw, const float* __restrict__ gb,
    const float* __restrict__ aw, const float* __restrict__ ab,
    _Float16* __restrict__ theta_h, float* __restrict__ phi_feat,
    float* __restrict__ g_feat, float* __restrict__ atten) {
  __shared__ float xt[64 * 64];   // [c][pw]
  __shared__ float wts[100 * 64];
  __shared__ float bs[100];
  int b = blockIdx.x;             // n*128 + ph
  int ph = b & 127, n = b >> 7;
  int t = threadIdx.x;
  for (int idx = t; idx < 4096; idx += 256) {
    int c = idx >> 6, q = idx & 63;
    xt[idx] = xd[(((size_t)(n * CH + c) * 128) + ph) * 64 + q];
  }
  for (int idx = t; idx < 6400; idx += 256) {
    int o = idx >> 6, c = idx & 63;
    float w;
    if (o < 32)      w = tw[o * 64 + c];
    else if (o < 64) w = pw[(o - 32) * 64 + c];
    else if (o < 96) w = gw[(o - 64) * 64 + c];
    else             w = aw[(o - 96) * 64 + c];
    wts[idx] = w;
  }
  if (t < 100)
    bs[t] = t < 32 ? tb[t] : t < 64 ? pb[t - 32] : t < 96 ? gb[t - 64] : ab[t - 96];
  __syncthreads();
  for (int idx = t; idx < 6400; idx += 256) {
    int o = idx % 100, q = idx / 100;
    float acc = bs[o];
#pragma unroll 8
    for (int c = 0; c < 64; ++c) acc += wts[o * 64 + c] * xt[c * 64 + q];
    int p = ph * 64 + q;
    if (o < 32)      theta_h[((size_t)n * P4 + p) * 32 + o] = (_Float16)acc;
    else if (o < 64) phi_feat[((size_t)(n * CHH) + (o - 32)) * P4 + p] = acc;
    else if (o < 96) g_feat[((size_t)(n * CHH) + (o - 64)) * P4 + p] = acc;
    else             atten[((size_t)(n * 4) + (o - 96)) * P4 + p] = acc;
  }
}

// ---------------- PAPA pyramid pooling -> phi_h (key-major), g_h (ch-major), f16, padded ----------------
__global__ __launch_bounds__(256) void k_papa(const float* __restrict__ phi_feat,
                                              const float* __restrict__ g_feat,
                                              const float* __restrict__ atten,
                                              _Float16* __restrict__ phi_h,
                                              _Float16* __restrict__ g_h) {
  __shared__ float red[256];
  int key = blockIdx.x, n = blockIdx.y, t = threadIdx.x;
  int slot = t & 63, rep = t >> 6;
  int c = slot & 31, which = slot >> 5;
  if (key >= NK) {                 // zero the padding keys 337..351
    if (t < 64) {
      if (which == 0) phi_h[((size_t)n * NKP + key) * 32 + c] = (_Float16)0.0f;
      else            g_h[((size_t)(n * CHH) + c) * NKP + key] = (_Float16)0.0f;
    }
    return;
  }
  int s, bin, br;
  if (key == 0)      { s = 1;  bin = 0;        br = 0; }
  else if (key < 17) { s = 4;  bin = key - 1;  br = 1; }
  else if (key < 81) { s = 8;  bin = key - 17; br = 2; }
  else               { s = 16; bin = key - 81; br = 3; }
  int rows = 128 / s, cols = 64 / s, cnt = rows * cols;
  int bh = bin / s, bw = bin % s;
  const float* fsrc = (which == 0 ? phi_feat : g_feat) + ((size_t)(n * CHH) + c) * P4;
  const float* asrc = atten + ((size_t)(n * 4) + br) * P4;
  float acc = 0.f;
  for (int it = rep; it < cnt; it += 4) {
    int r = it / cols, q = it % cols;
    int p = (bh * rows + r) * 64 + bw * cols + q;
    acc += fsrc[p] * asrc[p];
  }
  red[t] = acc; __syncthreads();
  if (rep == 0) {
    float v = (red[slot] + red[slot + 64] + red[slot + 128] + red[slot + 192]) / (float)cnt;
    if (which == 0) phi_h[((size_t)n * NKP + key) * 32 + c] = (_Float16)v;
    else            g_h[((size_t)(n * CHH) + c) * NKP + key] = (_Float16)v;
  }
}

// ---------------- fused attention: WMMA logits -> Ra/Rd/S softmax -> WMMA S@g -> z proj ----------------
__global__ __launch_bounds__(32) void k_attn(const _Float16* __restrict__ theta_h,
                                             const _Float16* __restrict__ phi_h,
                                             const _Float16* __restrict__ g_h,
                                             const float* __restrict__ depth1,
                                             const float* __restrict__ d2e,
                                             const float* __restrict__ s1,
                                             const float* __restrict__ dd1_b,
                                             const float* __restrict__ z_w,
                                             const float* __restrict__ z_b,
                                             float* __restrict__ z_out) {
  __shared__ float    ra[16 * NKP];     // logits, then S-pre-softmax values
  __shared__ _Float16 sS[16 * NKP];     // final S (f16, padded with 0)
  __shared__ float    yld[16 * 32];
  __shared__ float    kc_s1[NKP], kc_b1[NKP], kc_d2[NKP];
  __shared__ float    zw_l[64 * 32], zb_l[64];

  const int n = blockIdx.y;
  const int p0 = blockIdx.x * 16;
  const int L = threadIdx.x;
  const int m = L & 15;
  const int khalf = (L >> 4) << 3;
  const int mbase = (L >> 4) << 3;

  for (int idx = L; idx < NK; idx += 32) {
    kc_s1[idx] = s1[idx];
    kc_b1[idx] = dd1_b[idx];
    kc_d2[idx] = d2e[n * NK + idx];
  }
  for (int idx = L; idx < 64 * 32; idx += 32) zw_l[idx] = z_w[idx];
  for (int idx = L; idx < 64; idx += 32) zb_l[idx] = z_b[idx];

  // A fragment: theta rows p0..p0+15 (16x32 f16)
  union { v16h h; unsigned int u[8]; } A;
  const _Float16* trow = theta_h + ((size_t)n * P4 + p0 + m) * 32;
#pragma unroll
  for (int j = 0; j < 8; ++j) {
    int k = ((j & 3) << 1) + khalf + ((j >> 2) << 4);
    A.u[j] = *(const unsigned int*)(trow + k);
  }

  // logits = theta @ phi  (22 column tiles of 16 keys, K=32 per WMMA)
  for (int kt = 0; kt < NKP / 16; ++kt) {
    union { v16h h; unsigned int u[8]; } B;
    int key = kt * 16 + (L & 15);
    const _Float16* prow = phi_h + ((size_t)n * NKP + key) * 32;
#pragma unroll
    for (int j = 0; j < 8; ++j) {
      int k = ((j & 3) << 1) + khalf + ((j >> 2) << 4);
      B.u[j] = *(const unsigned int*)(prow + k);
    }
    v8f cacc = {};
    cacc = __builtin_amdgcn_wmma_f32_16x16x32_f16(false, A.h, false, B.h,
                                                  (short)0, cacc, false, false);
    int ncol = L & 15;
#pragma unroll
    for (int r = 0; r < 8; ++r)
      ra[(mbase + r) * NKP + kt * 16 + ncol] = cacc[r];
  }
  __syncthreads();

  // softmax chain: Ra=softmax(logits), Rd=softmax(min-ratio), S=softmax(Ra*Rd)
  if (L < 16) {
    const int mm = L;
    const float dp = depth1[n * P4 + p0 + mm];
    float mx1 = -1e30f, mx2 = -1e30f;
    for (int k = 0; k < NK; ++k) {
      mx1 = fmaxf(mx1, ra[mm * NKP + k]);
      float d1 = dp * kc_s1[k] + kc_b1[k];
      float d2v = kc_d2[k];
      float r = fminf(d1 / (d2v + EPSV), d2v / (d1 + EPSV));
      mx2 = fmaxf(mx2, r);
    }
    float sm1 = 0.f, sm2 = 0.f;
    for (int k = 0; k < NK; ++k) {
      sm1 += __expf(ra[mm * NKP + k] - mx1);
      float d1 = dp * kc_s1[k] + kc_b1[k];
      float d2v = kc_d2[k];
      float r = fminf(d1 / (d2v + EPSV), d2v / (d1 + EPSV));
      sm2 += __expf(r - mx2);
    }
    float inv12 = 1.0f / (sm1 * sm2);
    float mxt = -1e30f;
    for (int k = 0; k < NK; ++k) {
      float la = ra[mm * NKP + k];
      float d1 = dp * kc_s1[k] + kc_b1[k];
      float d2v = kc_d2[k];
      float r = fminf(d1 / (d2v + EPSV), d2v / (d1 + EPSV));
      float tt = __expf(la + r - mx1 - mx2) * inv12;   // Ra*Rd
      ra[mm * NKP + k] = tt;
      mxt = fmaxf(mxt, tt);
    }
    float smt = 0.f;
    for (int k = 0; k < NK; ++k) smt += __expf(ra[mm * NKP + k] - mxt);
    float invt = 1.0f / smt;
    for (int k = 0; k < NK; ++k)
      sS[mm * NKP + k] = (_Float16)(__expf(ra[mm * NKP + k] - mxt) * invt);
    for (int k = NK; k < NKP; ++k) sS[mm * NKP + k] = (_Float16)0.0f;
  }
  __syncthreads();

  // y = S @ g : K tiles of 32 keys, 2 column tiles of 16 channels
  v8f acc0 = {}, acc1 = {};
  for (int kt2 = 0; kt2 < NKP / 32; ++kt2) {
    union { v16h h; unsigned int u[8]; } A2;
#pragma unroll
    for (int j = 0; j < 8; ++j) {
      int k = kt2 * 32 + ((j & 3) << 1) + khalf + ((j >> 2) << 4);
      A2.u[j] = *(const unsigned int*)((const _Float16*)sS + m * NKP + k);
    }
    {
      union { v16h h; unsigned int u[8]; } B2;
      const _Float16* grow = g_h + ((size_t)(n * CHH) + (L & 15)) * NKP;
#pragma unroll
      for (int j = 0; j < 8; ++j) {
        int k = kt2 * 32 + ((j & 3) << 1) + khalf + ((j >> 2) << 4);
        B2.u[j] = *(const unsigned int*)(grow + k);
      }
      acc0 = __builtin_amdgcn_wmma_f32_16x16x32_f16(false, A2.h, false, B2.h,
                                                    (short)0, acc0, false, false);
    }
    {
      union { v16h h; unsigned int u[8]; } B2;
      const _Float16* grow = g_h + ((size_t)(n * CHH) + 16 + (L & 15)) * NKP;
#pragma unroll
      for (int j = 0; j < 8; ++j) {
        int k = kt2 * 32 + ((j & 3) << 1) + khalf + ((j >> 2) << 4);
        B2.u[j] = *(const unsigned int*)(grow + k);
      }
      acc1 = __builtin_amdgcn_wmma_f32_16x16x32_f16(false, A2.h, false, B2.h,
                                                    (short)0, acc1, false, false);
    }
  }
  {
    int ncol = L & 15;
#pragma unroll
    for (int r = 0; r < 8; ++r) {
      yld[(mbase + r) * 32 + ncol] = acc0[r];
      yld[(mbase + r) * 32 + 16 + ncol] = acc1[r];
    }
  }
  __syncthreads();

  // z_out[n, p, co] = z_b[co] + sum_c z_w[co,c] * y[p,c]   (ch-last for L2-friendly upsample)
  for (int i = 0; i < 32; ++i) {
    int mm = i >> 1;
    int co = ((i & 1) << 5) + L;
    float acc = zb_l[co];
#pragma unroll 8
    for (int c = 0; c < 32; ++c) acc += zw_l[co * 32 + c] * yld[mm * 32 + c];
    z_out[((size_t)n * P4 + p0 + mm) * 64 + co] = acc;
  }
}

// ---------------- out = x + bilinear_up(z, half-pixel, 4x) ----------------
__global__ __launch_bounds__(256) void k_up(const float* __restrict__ x,
                                            const float* __restrict__ z,
                                            float* __restrict__ out) {
  int gid = blockIdx.x * 256 + threadIdx.x;   // 16,777,216 quads
  int wq = gid & 63;
  int h = (gid >> 6) & 511;
  int c = (gid >> 15) & 63;
  int n = gid >> 21;
  float sh = fmaxf((h + 0.5f) * 0.25f - 0.5f, 0.0f);
  int lh = imin((int)sh, 127);
  int hh2 = imin(lh + 1, 127);
  float fh = sh - (float)lh;
  const float* zn = z + (size_t)n * P4 * 64;
  size_t base = (((size_t)(n * CH) + c) * 512 + h) * 256 + (size_t)wq * 4;
  float4 xv = *(const float4*)(x + base);
  float res[4];
#pragma unroll
  for (int i = 0; i < 4; ++i) {
    int w = wq * 4 + i;
    float sw = fmaxf((w + 0.5f) * 0.25f - 0.5f, 0.0f);
    int lw = imin((int)sw, 63);
    int hw2 = imin(lw + 1, 63);
    float fw = sw - (float)lw;
    float v00 = zn[(lh * 64 + lw) * 64 + c];
    float v01 = zn[(lh * 64 + hw2) * 64 + c];
    float v10 = zn[(hh2 * 64 + lw) * 64 + c];
    float v11 = zn[(hh2 * 64 + hw2) * 64 + c];
    res[i] = v00 * (1 - fh) * (1 - fw) + v01 * (1 - fh) * fw
           + v10 * fh * (1 - fw)       + v11 * fh * fw;
  }
  float4 ov;
  ov.x = xv.x + res[0]; ov.y = xv.y + res[1];
  ov.z = xv.z + res[2]; ov.w = xv.w + res[3];
  *(float4*)(out + base) = ov;
}

extern "C" void kernel_launch(void* const* d_in, const int* in_sizes, int n_in,
                              void* d_out, int out_size, void* d_ws, size_t ws_size,
                              hipStream_t stream) {
  const float* x        = (const float*)d_in[0];
  const float* dm       = (const float*)d_in[1];
  const float* theta_w  = (const float*)d_in[2];
  const float* theta_b  = (const float*)d_in[3];
  const float* phi_w    = (const float*)d_in[4];
  const float* phi_b    = (const float*)d_in[5];
  const float* g_w      = (const float*)d_in[6];
  const float* g_b      = (const float*)d_in[7];
  const float* atten_w  = (const float*)d_in[8];
  const float* atten_b  = (const float*)d_in[9];
  const float* dd1_w    = (const float*)d_in[10];
  const float* dd1_b    = (const float*)d_in[11];
  const float* dd2_w    = (const float*)d_in[12];
  const float* dd2_b    = (const float*)d_in[13];
  const float* down_w   = (const float*)d_in[14];
  const float* z_w      = (const float*)d_in[15];
  const float* z_b      = (const float*)d_in[16];
  float* out = (float*)d_out;

  // workspace layout (float units); x_down aliased with z_out (disjoint lifetimes)
  float* ws = (float*)d_ws;
  float*    x_down  = ws;                               // 4,194,304 f
  float*    z_out   = ws;                               // alias
  _Float16* theta_h = (_Float16*)(ws + 4194304);        // 1,048,576 f
  float*    phi_ft  = ws + 5242880;                     // 2,097,152 f
  float*    g_ft    = ws + 7340032;                     // 2,097,152 f
  float*    atten   = ws + 9437184;                     //   262,144 f
  _Float16* phi_h   = (_Float16*)(ws + 9699328);        //    45,056 f
  _Float16* g_h     = (_Float16*)(ws + 9744384);        //    45,056 f
  float*    depth1  = ws + 9789440;                     //    65,536 f
  float*    d2e     = ws + 9854976;                     //     4,096 f
  float*    s1      = ws + 9859072;                     //       512 f

  k_s1<<<NK, 256, 0, stream>>>(dd1_w, s1);
  k_d2e<<<NB, 256, 0, stream>>>(dm, dd2_w, dd2_b, d2e);
  k_depth1<<<(NB * P4) / 256, 256, 0, stream>>>(dm, depth1);
  k_down<<<NB * CH * 128, 256, 0, stream>>>(x, down_w, x_down);
  k_proj<<<NB * 128, 256, 0, stream>>>(x_down, theta_w, theta_b, phi_w, phi_b,
                                       g_w, g_b, atten_w, atten_b,
                                       theta_h, phi_ft, g_ft, atten);
  k_papa<<<dim3(NKP, NB), 256, 0, stream>>>(phi_ft, g_ft, atten, phi_h, g_h);
  k_attn<<<dim3(P4 / 16, NB), 32, 0, stream>>>(theta_h, phi_h, g_h, depth1, d2e,
                                               s1, dd1_b, z_w, z_b, z_out);
  k_up<<<(NB * CH * 512 * 256 / 4) / 256, 256, 0, stream>>>(x, z_out, out);
}
